// FP4Linear_53309134078227
// MI455X (gfx1250) — 8-bit matrix operand order repaired and checked by reading, compiled, not run on hardware
//
#include <hip/hip_runtime.h>
#include <stdint.h>


typedef int      v2i   __attribute__((ext_vector_type(2)));
typedef int      v4i   __attribute__((ext_vector_type(4)));
typedef int      v16i  __attribute__((ext_vector_type(16)));
typedef float    v4f   __attribute__((ext_vector_type(4)));
typedef float    v8f   __attribute__((ext_vector_type(8)));
typedef unsigned u32x4 __attribute__((ext_vector_type(4)));
typedef int      i32x4 __attribute__((ext_vector_type(4)));
typedef int      i32x8 __attribute__((ext_vector_type(8)));

#define KDIM 4096
#define NDIM 4096
#define BM   64
#define BN   128
#define BK   128
#define LDSS 144

#if __has_builtin(__builtin_amdgcn_tensor_load_to_lds) && __has_builtin(__builtin_amdgcn_s_wait_tensorcnt)
#define USE_TDM 1
#else
#define USE_TDM 0
#endif

static __device__ __forceinline__ unsigned umaxu(unsigned a, unsigned b) { return a > b ? a : b; }

static __device__ __forceinline__ unsigned fp8_of(float t, float inv_scale) {
  float a = __builtin_fabsf(t) * inv_scale;
  unsigned c = 0;
  c += (a >= 0.25f) ? 0x30u : 0u;
  c += (a >= 0.75f) ? 8u : 0u;
  c += (a >= 1.25f) ? 4u : 0u;
  c += (a >= 1.75f) ? 4u : 0u;
  c += (a >= 2.50f) ? 4u : 0u;
  c += (a >= 3.50f) ? 4u : 0u;
  c += (a >= 5.00f) ? 4u : 0u;
  c |= (__float_as_uint(t) >> 24) & 0x80u;
  return c;
}

static __device__ __forceinline__ float inv_scale_of(unsigned amax_bits) {
  return 1.0f / fmaxf(__uint_as_float(amax_bits) * (1.0f / 6.0f), 1e-12f);
}

__global__ void init_amax_kernel(unsigned* a) {
  if (threadIdx.x < 2) a[threadIdx.x] = 0u;
}

__global__ void absmax_kernel(const v4f* __restrict__ p, long n4,
                              unsigned* __restrict__ out) {
  __shared__ unsigned red[256];
  unsigned m = 0;
  for (long i = (long)blockIdx.x * blockDim.x + threadIdx.x; i < n4;
       i += (long)gridDim.x * blockDim.x) {
    v4f v = __builtin_nontemporal_load(&p[i]);
    m = umaxu(m, __float_as_uint(v.x) & 0x7fffffffu);
    m = umaxu(m, __float_as_uint(v.y) & 0x7fffffffu);
    m = umaxu(m, __float_as_uint(v.z) & 0x7fffffffu);
    m = umaxu(m, __float_as_uint(v.w) & 0x7fffffffu);
  }
  red[threadIdx.x] = m;
  __syncthreads();
  for (int s = 128; s > 0; s >>= 1) {
    if ((int)threadIdx.x < s) red[threadIdx.x] = umaxu(red[threadIdx.x], red[threadIdx.x + s]);
    __syncthreads();
  }
  if (threadIdx.x == 0) atomicMax(out, red[0]);
}

__global__ void quant_x_kernel(const float* __restrict__ x, unsigned char* __restrict__ qx,
                               const unsigned* __restrict__ amax_bits, long n) {
  const float inv = inv_scale_of(*amax_bits);
  long i = ((long)blockIdx.x * blockDim.x + threadIdx.x) * 8;
  if (i + 8 > n) return;
  v4f a = __builtin_nontemporal_load((const v4f*)(x + i));
  v4f b = __builtin_nontemporal_load((const v4f*)(x + i + 4));
  uint2 pk;
  pk.x = fp8_of(a.x, inv) | (fp8_of(a.y, inv) << 8) |
         (fp8_of(a.z, inv) << 16) | (fp8_of(a.w, inv) << 24);
  pk.y = fp8_of(b.x, inv) | (fp8_of(b.y, inv) << 8) |
         (fp8_of(b.z, inv) << 16) | (fp8_of(b.w, inv) << 24);
  *(uint2*)(qx + i) = pk;
}

__global__ void quant_wT_kernel(const float* __restrict__ w, unsigned char* __restrict__ qwT,
                                const unsigned* __restrict__ amax_bits) {
  const float inv = inv_scale_of(*amax_bits);
  const int  t  = threadIdx.x;
  const long n  = (long)blockIdx.x * 64 + (t & 63);
  const long k0 = (long)blockIdx.y * 16 + (t >> 6) * 4;
  unsigned pk = 0;
  #pragma unroll
  for (int j = 0; j < 4; ++j) {
    float v = __builtin_nontemporal_load(&w[(k0 + j) * (long)NDIM + n]);
    pk |= fp8_of(v, inv) << (8 * j);
  }
  *(unsigned*)(qwT + n * (long)KDIM + k0) = pk;
}

#if USE_TDM
static __device__ __forceinline__ unsigned lds_off(const void* p) {
  return (unsigned)(unsigned long long)(size_t)p;
}

static __device__ __forceinline__ void tdm_load_2d(unsigned lds_addr,
                                                   unsigned long long gaddr,
                                                   unsigned tile_rows,
                                                   unsigned tensor_rows) {
  u32x4 g0;
  g0.x = 1u;
  g0.y = lds_addr;
  g0.z = (unsigned)gaddr;
  g0.w = ((unsigned)(gaddr >> 32) & 0x01FFFFFFu)
         | (2u << 30);
  i32x8 g1;
  g1[0] = (int)((1u << 20) | (4u << 22) | (3u << 25));
  g1[1] = (int)(((unsigned)KDIM & 0xFFFFu) << 16);
  g1[2] = (int)((((unsigned)KDIM >> 16) & 0xFFFFu) |
                ((tensor_rows & 0xFFFFu) << 16));
  g1[3] = (int)(((tensor_rows >> 16) & 0xFFFFu) |
                (((unsigned)BK & 0xFFFFu) << 16));
  g1[4] = (int)(tile_rows & 0xFFFFu);
  g1[5] = (int)KDIM;
  g1[6] = 0;
  g1[7] = 0;
  i32x4 z4 = {0, 0, 0, 0};
  i32x8 z8 = {0, 0, 0, 0, 0, 0, 0, 0};
  __builtin_amdgcn_tensor_load_to_lds(g0, g1, z4, z4, z8, 0);
}
#endif

static __device__ __forceinline__ void do_tile_mma(const unsigned char* ldsA,
                                                   const unsigned char* ldsB,
                                                   int waveM, int waveN, int l16, int hi,
                                                   v8f (&acc)[2][2]) {
  v16i af[2], bf[2];
  #pragma unroll
  for (int mt = 0; mt < 2; ++mt) {
    const unsigned char* p = &ldsA[(waveM * 32 + mt * 16 + l16) * LDSS + hi * 16];
    #pragma unroll
    for (int c = 0; c < 8; ++c) {
      v2i t = *(const v2i*)(p + ((c >> 1) * 32 + (c & 1) * 8));
      af[mt][2 * c]     = t.x;
      af[mt][2 * c + 1] = t.y;
    }
  }
  #pragma unroll
  for (int nt = 0; nt < 2; ++nt) {
    const unsigned char* p = &ldsB[(waveN * 32 + nt * 16 + l16) * LDSS + hi * 16];
    #pragma unroll
    for (int q = 0; q < 4; ++q) {
      v4i t = *(const v4i*)(p + q * 32);
      bf[nt][4 * q]     = t.x;
      bf[nt][4 * q + 1] = t.y;
      bf[nt][4 * q + 2] = t.z;
      bf[nt][4 * q + 3] = t.w;
    }
  }
  #pragma unroll
  for (int mt = 0; mt < 2; ++mt)
    #pragma unroll
    for (int nt = 0; nt < 2; ++nt)
      acc[mt][nt] = __builtin_amdgcn_wmma_f32_16x16x128_fp8_fp8(
          af[mt], bf[nt], (short)0, acc[mt][nt], false, false);
}

__global__ __launch_bounds__(256) void fp4gemm_kernel(
    const unsigned char* __restrict__ qx, const unsigned char* __restrict__ qwT,
    const float* __restrict__ bias, const unsigned* __restrict__ amax_bits,
    float* __restrict__ out) {
  const int tid   = threadIdx.x;
  const int lane  = tid & 31;
  const int wave  = tid >> 5;
  const int waveM = wave >> 2;
  const int waveN = wave & 3;
  const int l16   = lane & 15;
  const int hi    = lane >> 4;

  const long mBase = (long)blockIdx.y * BM;
  const long nBase = (long)blockIdx.x * BN;

  v8f acc[2][2];
  #pragma unroll
  for (int i = 0; i < 2; ++i)
    #pragma unroll
    for (int j = 0; j < 2; ++j) acc[i][j] = (v8f){0, 0, 0, 0, 0, 0, 0, 0};

  const int NK = KDIM / BK;

#if USE_TDM
  __shared__ __align__(128) unsigned char smem[2 * BM * LDSS + 2 * BN * LDSS];
  const unsigned ASZ = BM * LDSS;
  const unsigned BSZ = BN * LDSS;
  const unsigned ldsBase = lds_off(&smem[0]);

  const unsigned long long aAddr =
      (unsigned long long)(size_t)qx + (unsigned long long)mBase * KDIM;
  const unsigned long long bAddr =
      (unsigned long long)(size_t)qwT + (unsigned long long)nBase * KDIM;
  const unsigned Mrows = gridDim.y * BM;

  if (wave == 0)      tdm_load_2d(ldsBase + 0u,        aAddr, BM, Mrows);
  else if (wave == 1) tdm_load_2d(ldsBase + 2u * ASZ,  bAddr, BN, NDIM);
  if (wave < 2) __builtin_amdgcn_s_wait_tensorcnt(0);
  __syncthreads();

  unsigned cur = 0;
  for (int kt = 0; kt < NK; ++kt) {
    const bool more = (kt + 1) < NK;
    const unsigned aOffC = cur * ASZ;
    const unsigned bOffC = 2u * ASZ + cur * BSZ;
    if (more) {
      const unsigned long long ko = (unsigned long long)(kt + 1) * BK;
      const unsigned nxt = cur ^ 1u;
      if (wave == 0)      tdm_load_2d(ldsBase + nxt * ASZ,            aAddr + ko, BM, Mrows);
      else if (wave == 1) tdm_load_2d(ldsBase + 2u * ASZ + nxt * BSZ, bAddr + ko, BN, NDIM);
    }

    do_tile_mma(smem + aOffC, smem + bOffC, waveM, waveN, l16, hi, acc);

    if (more && wave < 2) __builtin_amdgcn_s_wait_tensorcnt(0);
    __syncthreads();
    cur ^= 1u;
  }
#else
  __shared__ __align__(16) unsigned char ldsA[BM * LDSS];
  __shared__ __align__(16) unsigned char ldsB[BN * LDSS];

  const int arow = tid >> 2, akoff = (tid & 3) * 32;
  const int brow = tid >> 1, bkoff = (tid & 1) * 64;
  const unsigned char* aG = qx  + (mBase + arow) * (long)KDIM + akoff;
  const unsigned char* bG = qwT + (nBase + brow) * (long)KDIM + bkoff;

  v4i aReg[2], bReg[4];
  aReg[0] = *(const v4i*)(aG + 0);
  aReg[1] = *(const v4i*)(aG + 16);
  #pragma unroll
  for (int i = 0; i < 4; ++i) bReg[i] = *(const v4i*)(bG + 16 * i);

  *(v4i*)(&ldsA[arow * LDSS + akoff + 0])  = aReg[0];
  *(v4i*)(&ldsA[arow * LDSS + akoff + 16]) = aReg[1];
  #pragma unroll
  for (int i = 0; i < 4; ++i) *(v4i*)(&ldsB[brow * LDSS + bkoff + 16 * i]) = bReg[i];
  __syncthreads();

  for (int kt = 0; kt < NK; ++kt) {
    const bool more = (kt + 1) < NK;
    if (more) {
      const long ko = (long)(kt + 1) * BK;
      aReg[0] = *(const v4i*)(aG + ko + 0);
      aReg[1] = *(const v4i*)(aG + ko + 16);
      #pragma unroll
      for (int i = 0; i < 4; ++i) bReg[i] = *(const v4i*)(bG + ko + 16 * i);
    }

    do_tile_mma(ldsA, ldsB, waveM, waveN, l16, hi, acc);

    __syncthreads();
    if (more) {
      *(v4i*)(&ldsA[arow * LDSS + akoff + 0])  = aReg[0];
      *(v4i*)(&ldsA[arow * LDSS + akoff + 16]) = aReg[1];
      #pragma unroll
      for (int i = 0; i < 4; ++i) *(v4i*)(&ldsB[brow * LDSS + bkoff + 16 * i]) = bReg[i];
    }
    __syncthreads();
  }
#endif

  const float sx = fmaxf(__uint_as_float(amax_bits[0]) * (1.0f / 6.0f), 1e-12f);
  const float sw = fmaxf(__uint_as_float(amax_bits[1]) * (1.0f / 6.0f), 1e-12f);
  const float s  = sx * sw;

  #pragma unroll
  for (int mt = 0; mt < 2; ++mt) {
    const long m0 = mBase + waveM * 32 + mt * 16 + hi * 8;
    #pragma unroll
    for (int nt = 0; nt < 2; ++nt) {
      const long  n  = nBase + waveN * 32 + nt * 16 + l16;
      const float bv = bias[n];
      #pragma unroll
      for (int v = 0; v < 8; ++v)
        __builtin_nontemporal_store(acc[mt][nt][v] * s + bv,
                                    &out[(m0 + v) * (long)NDIM + n]);
    }
  }
}

extern "C" void kernel_launch(void* const* d_in, const int* in_sizes, int n_in,
                              void* d_out, int out_size, void* d_ws, size_t ws_size,
                              hipStream_t stream) {
  (void)n_in; (void)out_size; (void)ws_size;
  const float* x    = (const float*)d_in[0];
  const float* w    = (const float*)d_in[1];
  const float* bias = (const float*)d_in[2];
  float*       out  = (float*)d_out;

  const long xN = (long)in_sizes[0];
  const long wN = (long)in_sizes[1];
  const int  M  = (int)(xN / KDIM);

  unsigned char* ws  = (unsigned char*)d_ws;
  unsigned*      am  = (unsigned*)ws;
  unsigned char* qx  = ws + 64;
  unsigned char* qwT = qx + xN;

  init_amax_kernel<<<1, 32, 0, stream>>>(am);
  absmax_kernel<<<4096, 256, 0, stream>>>((const v4f*)x, xN / 4, am + 0);
  absmax_kernel<<<1024, 256, 0, stream>>>((const v4f*)w, wN / 4, am + 1);

  quant_x_kernel<<<(unsigned)(xN / 8 / 256), 256, 0, stream>>>(x, qx, am + 0, xN);
  dim3 gw(NDIM / 64, KDIM / 16);
  quant_wT_kernel<<<gw, 256, 0, stream>>>(w, qwT, am + 1);

  dim3 gg(NDIM / BN, M / BM);
  fp4gemm_kernel<<<gg, 256, 0, stream>>>(qx, qwT, bias, am, out);
}
